// SARAN_89575837925919
// MI455X (gfx1250) — matrix operand order repaired and checked by reading, not compiled, not run on hardware
//
#include <hip/hip_runtime.h>
#include <cmath>

typedef __bf16 bf16;
typedef __attribute__((ext_vector_type(8)))  __bf16 v8bf;
typedef __attribute__((ext_vector_type(16))) __bf16 v16bf;
typedef __attribute__((ext_vector_type(8)))  float  v8f;

#define LDT 40  // LDS tile row stride in elements (80B: keeps 16B chunks aligned, spreads banks)

// ---------------------------------------------------------------- converts
__global__ __launch_bounds__(256) void f32_to_bf16_kernel(const float* __restrict__ src,
                                                          bf16* __restrict__ dst, long n) {
    for (long i = (long)blockIdx.x * blockDim.x + threadIdx.x; i < n;
         i += (long)gridDim.x * blockDim.x)
        dst[i] = (bf16)src[i];
}

// ---------------------------------------------------------------- embedding
__global__ __launch_bounds__(256) void embed_kernel(const int* __restrict__ idx,
                                                    const float* __restrict__ we,
                                                    const float* __restrict__ wp,
                                                    bf16* __restrict__ x,
                                                    int T, int D, long total) {
    long i = (long)blockIdx.x * blockDim.x + threadIdx.x;
    if (i >= total) return;
    long row = i / D;
    int  d   = (int)(i - row * D);
    int  t   = (int)(row % T);
    x[i] = (bf16)(we[(size_t)idx[row] * D + d] + wp[(size_t)t * D + d]);
}

// ---------------------------------------------------------------- softmax (causal), T=256
__global__ __launch_bounds__(256) void softmax_kernel(const float* __restrict__ scores,
                                                      bf16* __restrict__ probs) {
    __shared__ float red[256];
    int row = blockIdx.x;            // row = b*256 + t
    int t   = row & 255;
    int s   = threadIdx.x;
    float v = scores[(size_t)row * 256 + s];
    if (s > t) v = -INFINITY;        // causal mask
    red[s] = v; __syncthreads();
    for (int off = 128; off > 0; off >>= 1) {
        if (s < off) red[s] = fmaxf(red[s], red[s + off]);
        __syncthreads();
    }
    float m = red[0]; __syncthreads();
    float e = __expf(v - m);
    red[s] = e; __syncthreads();
    for (int off = 128; off > 0; off >>= 1) {
        if (s < off) red[s] += red[s + off];
        __syncthreads();
    }
    float sum = red[0];
    probs[(size_t)row * 256 + s] = (bf16)(e / sum);
}

// ---------------------------------------------------------------- WMMA fragment loaders
// A frag (16x32 bf16): lane L -> M=L&15; elem e -> K=(L>>4)*8 + (e>>3)*16 + (e&7)
__device__ __forceinline__ v16bf load_a_frag(const bf16* As, int row, int half) {
    const v8bf lo = *(const v8bf*)(As + row * LDT + half * 8);
    const v8bf hi = *(const v8bf*)(As + row * LDT + half * 8 + 16);
    v16bf a;
#pragma unroll
    for (int i = 0; i < 8; ++i) { a[i] = lo[i]; a[i + 8] = hi[i]; }
    return a;
}
// B frag (32x16 bf16): lane L -> N=L&15; elem e -> K=(L>>4)*8 + (e>>3)*16 + (e&7)   (LDS tile is [n][k])
__device__ __forceinline__ v16bf load_b_frag(const bf16* Bs, int n, int half) {
    const v8bf lo = *(const v8bf*)(Bs + n * LDT + half * 8);
    const v8bf hi = *(const v8bf*)(Bs + n * LDT + half * 8 + 16);
    v16bf b;
#pragma unroll
    for (int i = 0; i < 8; ++i) { b[i] = lo[i]; b[i + 8] = hi[i]; }
    return b;
}

// ---------------------------------------------------------------- WMMA GEMM
// C[m,n] = sum_k A[m,k] * Bsrc(k,n)
//   BKM=false : Bsrc memory is [N][K] (weight rows / k^T), k-contiguous
//   BKM=true  : Bsrc memory is [K][N] (V matrix), n-contiguous (transposed into LDS)
// OUTM: 0 = bf16 store, 1 = f32 * scale (scores), 2 = f32 + bias (logits)
// Block = 256 threads = 8 waves; workgroup tile 128(M) x 128(N); wave tile 64x32.
// Register-staged, LDS double-buffered: one barrier per k-step, global loads for
// step s+1 issued before the 8 WMMAs of step s (latency hidden under compute).
template <bool BKM, int OUTM>
__global__ __launch_bounds__(256) void gemm_wmma_kernel(
    const bf16* __restrict__ A, const bf16* __restrict__ B, void* __restrict__ Cv,
    const float* __restrict__ bias, int M, int N, int K,
    int lda, int ldb, int ldc, long sA, long sB, long sC, float scale) {
    __shared__ bf16 As[2][128 * LDT];   // 128(m) x 32(k), [m][k]
    __shared__ bf16 Bs[2][128 * LDT];   // 128(n) x 32(k), [n][k]

    const int tid = threadIdx.x;
    const int bm  = blockIdx.y * 128;
    const int bn  = blockIdx.x * 128;
    const int bz  = blockIdx.z;
    A += (size_t)bz * sA;
    B += (size_t)bz * sB;

    const int lane = tid & 31;
    const int w    = tid >> 5;
    const int wm   = (w >> 2) * 64;   // 2 waves along M (64 rows each)
    const int wn   = (w & 3) * 32;    // 4 waves along N (32 cols each)
    const int half = lane >> 4;
    const int ln   = lane & 15;

    // staging coords
    const int ar = tid >> 2, ac = (tid & 3) * 8;       // A / B(kN-major=false) pattern
    const int kk = tid >> 4, nc = (tid & 15) * 8;      // B(kN-major=true) pattern

    v8f acc[4][2] = {};
    v8bf sa0, sa1, sb0, sb1;

    const int nsteps = K >> 5;  // K / 32

    auto load_globals = [&](int s) {
        const int k0 = s << 5;
        sa0 = *(const v8bf*)(A + (size_t)(bm + ar) * lda + k0 + ac);
        sa1 = *(const v8bf*)(A + (size_t)(bm + ar + 64) * lda + k0 + ac);
        if (!BKM) {
            int gn0 = bn + ar;      if (gn0 > N - 1) gn0 = N - 1;
            int gn1 = bn + ar + 64; if (gn1 > N - 1) gn1 = N - 1;
            sb0 = *(const v8bf*)(B + (size_t)gn0 * ldb + k0 + ac);
            sb1 = *(const v8bf*)(B + (size_t)gn1 * ldb + k0 + ac);
        } else {
            sb0 = *(const v8bf*)(B + (size_t)(k0 + kk) * ldb + bn + nc);
            sb1 = *(const v8bf*)(B + (size_t)(k0 + kk + 16) * ldb + bn + nc);
        }
    };
    auto store_lds = [&](int buf) {
        *(v8bf*)(&As[buf][ar * LDT + ac])        = sa0;
        *(v8bf*)(&As[buf][(ar + 64) * LDT + ac]) = sa1;
        if (!BKM) {
            *(v8bf*)(&Bs[buf][ar * LDT + ac])        = sb0;
            *(v8bf*)(&Bs[buf][(ar + 64) * LDT + ac]) = sb1;
        } else {
#pragma unroll
            for (int j = 0; j < 8; ++j) {
                Bs[buf][(nc + j) * LDT + kk]      = sb0[j];
                Bs[buf][(nc + j) * LDT + kk + 16] = sb1[j];
            }
        }
    };

    load_globals(0);
    store_lds(0);

    for (int s = 0; s < nsteps; ++s) {
        __syncthreads();
        const int buf = s & 1;
        if (s + 1 < nsteps) load_globals(s + 1);   // issue loads; waited at store_lds below

        v16bf af[4], bfg[2];
#pragma unroll
        for (int i = 0; i < 4; ++i) af[i] = load_a_frag(&As[buf][0], wm + i * 16 + ln, half);
#pragma unroll
        for (int j = 0; j < 2; ++j) bfg[j] = load_b_frag(&Bs[buf][0], wn + j * 16 + ln, half);
#pragma unroll
        for (int i = 0; i < 4; ++i)
#pragma unroll
            for (int j = 0; j < 2; ++j)
                acc[i][j] = __builtin_amdgcn_wmma_f32_16x16x32_bf16(
                    false, af[i], false, bfg[j], (short)0, acc[i][j], false, false);

        if (s + 1 < nsteps) store_lds(buf ^ 1);
    }

    // --- store: C/D layout: VGPR r -> M = r + half*8, N = ln
#pragma unroll
    for (int i = 0; i < 4; ++i)
#pragma unroll
        for (int j = 0; j < 2; ++j) {
            int n = bn + wn + j * 16 + ln;
            if (n >= N) continue;
#pragma unroll
            for (int r = 0; r < 8; ++r) {
                int m = bm + wm + i * 16 + half * 8 + r;
                size_t cidx = (size_t)bz * sC + (size_t)m * ldc + n;
                float v = acc[i][j][r];
                if (OUTM == 0)      ((bf16*)Cv)[cidx]  = (bf16)v;
                else if (OUTM == 1) ((float*)Cv)[cidx] = v * scale;
                else                ((float*)Cv)[cidx] = v + bias[n];
            }
        }
}

// ---------------------------------------------------------------- launcher
extern "C" void kernel_launch(void* const* d_in, const int* in_sizes, int n_in,
                              void* d_out, int out_size, void* d_ws, size_t ws_size,
                              hipStream_t stream) {
    constexpr int Bb = 16, T = 256, D = 640, V = 50257;
    constexpr long NT = (long)Bb * T;          // 4096 rows

    const int*   idx   = (const int*)d_in[0];
    const float* w_emb = (const float*)d_in[1];
    const float* w_pos = (const float*)d_in[2];
    const float* wq    = (const float*)d_in[3];
    const float* wk    = (const float*)d_in[4];
    const float* wv    = (const float*)d_in[5];
    const float* w_out = (const float*)d_in[6];
    const float* b_out = (const float*)d_in[7];
    float* logits = (float*)d_out;

    // workspace carve-up (all chunk sizes multiples of 256B)
    uint8_t* ws = (uint8_t*)d_ws;
    const size_t XB = (size_t)NT * D * 2;        // bf16 activations
    const size_t WB = (size_t)D * D * 2;         // bf16 weight
    const size_t OB = (size_t)V * D * 2;         // bf16 w_out
    const size_t SB = (size_t)Bb * T * T * 4;    // f32 scores
    const size_t PB = (size_t)Bb * T * T * 2;    // bf16 probs
    bf16* x_bf   = (bf16*)(ws);                   size_t off = XB;
    bf16* wq_bf  = (bf16*)(ws + off); off += WB;
    bf16* wk_bf  = (bf16*)(ws + off); off += WB;
    bf16* wv_bf  = (bf16*)(ws + off); off += WB;
    bf16* wo_bf  = (bf16*)(ws + off); off += OB;
    bf16* q_bf   = (bf16*)(ws + off); off += XB;
    bf16* k_bf   = (bf16*)(ws + off); off += XB;
    bf16* v_bf   = (bf16*)(ws + off); off += XB;
    float* sc    = (float*)(ws + off); off += SB;
    bf16* p_bf   = (bf16*)(ws + off); off += PB;
    bf16* ao_bf  = (bf16*)(ws + off); off += XB;
    (void)ws_size; (void)in_sizes; (void)n_in; (void)out_size;

    // 1) weight converts
    f32_to_bf16_kernel<<<1024, 256, 0, stream>>>(wq,    wq_bf, (long)D * D);
    f32_to_bf16_kernel<<<1024, 256, 0, stream>>>(wk,    wk_bf, (long)D * D);
    f32_to_bf16_kernel<<<1024, 256, 0, stream>>>(wv,    wv_bf, (long)D * D);
    f32_to_bf16_kernel<<<4096, 256, 0, stream>>>(w_out, wo_bf, (long)V * D);

    // 2) embedding
    {
        long total = NT * D;
        embed_kernel<<<(unsigned)((total + 255) / 256), 256, 0, stream>>>(
            idx, w_emb, w_pos, x_bf, T, D, total);
    }

    // 3) Q/K/V projections: (4096x640) @ W[N][K]  -> bf16
    {
        dim3 g(D / 128, NT / 128, 1);
        gemm_wmma_kernel<false, 0><<<g, 256, 0, stream>>>(
            x_bf, wq_bf, q_bf, nullptr, (int)NT, D, D, D, D, D, 0, 0, 0, 1.0f);
        gemm_wmma_kernel<false, 0><<<g, 256, 0, stream>>>(
            x_bf, wk_bf, k_bf, nullptr, (int)NT, D, D, D, D, D, 0, 0, 0, 1.0f);
        gemm_wmma_kernel<false, 0><<<g, 256, 0, stream>>>(
            x_bf, wv_bf, v_bf, nullptr, (int)NT, D, D, D, D, D, 0, 0, 0, 1.0f);
    }

    // 4) scores = q @ k^T / sqrt(D), batched over 16  (B memory [N=s][K=d])
    {
        dim3 g(T / 128, T / 128, Bb);
        float scale = 1.0f / sqrtf((float)D);
        gemm_wmma_kernel<false, 1><<<g, 256, 0, stream>>>(
            q_bf, k_bf, sc, nullptr, T, T, D, D, D, T,
            (long)T * D, (long)T * D, (long)T * T, scale);
    }

    // 5) causal softmax -> probs bf16
    softmax_kernel<<<(unsigned)NT, 256, 0, stream>>>(sc, p_bf);

    // 6) attn_out = P @ V, batched  (B memory [K=s][N=d] -> transposed staging)
    {
        dim3 g(D / 128, T / 128, Bb);
        gemm_wmma_kernel<true, 0><<<g, 256, 0, stream>>>(
            p_bf, v_bf, ao_bf, nullptr, T, D, T, T, D, D,
            (long)T * T, (long)T * D, (long)T * D, 1.0f);
    }

    // 7) logits = attn_out @ w_out^T + b_out   (dominant GEMM, f32 out)
    {
        dim3 g((V + 127) / 128, NT / 128, 1);
        gemm_wmma_kernel<false, 2><<<g, 256, 0, stream>>>(
            ao_bf, wo_bf, logits, b_out, (int)NT, V, D, D, D, V, 0, 0, 0, 1.0f);
    }
}
